// ModulatedDeformConv2dFastPack_12996571038286
// MI455X (gfx1250) — matrix operand order repaired and checked by reading, not compiled, not run on hardware
//
#include <hip/hip_runtime.h>
#include <math.h>

// ModulatedDeformConv2d fused forward for gfx1250 (MI455X).
// Stage 1: offset-predictor conv (256->27, 3x3 SAME)  [VALU + LDS-staged weights]
// Stage 2: fused bilinear sampling (bf16 into LDS) + 1x1 conv GEMM via
//          v_wmma_f32_16x16x32_bf16 (M=BHW=32768, N=256, K=2304), f32 accum.
// Round-2 fix: k-loop unrolled x2 with ping-pong B buffers (no rotate copies,
// no WMMA->VALU WAR nops) and dual A-fragment sets for ds/WMMA overlap.

#define B_   8
#define C_   256
#define H_   64
#define W_   64
#define OC_  256
#define KK_  9
#define KD_  2304            // C_*KK_
#define HW_  4096            // H_*W_
#define NPKK (B_ * KK_ * HW_)  // 294912 floats per field

typedef __attribute__((ext_vector_type(16))) __bf16 v16bf;
typedef __attribute__((ext_vector_type(8)))  float  v8f;

static __device__ __forceinline__ unsigned short f2bf(float f) {
  union { float f; unsigned u; } v; v.f = f;
  unsigned r = v.u + 0x7FFFu + ((v.u >> 16) & 1u);   // round-to-nearest-even
  return (unsigned short)(r >> 16);
}

// ---------------------------------------------------------------- kernel 1 --
__global__ void k_wcvt(const float* __restrict__ w, unsigned short* __restrict__ wb, int n) {
  int i = blockIdx.x * blockDim.x + threadIdx.x;
  if (i < n) wb[i] = f2bf(w[i]);
}

// ---------------------------------------------------------------- kernel 2 --
// One thread per output pixel; 27 accumulators; weights staged in LDS in
// 16-input-channel chunks (27*16*9 floats = 15.2 KB).
__global__ __launch_bounds__(256)
void k_offsets(const float* __restrict__ x, const float* __restrict__ w_off,
               const float* __restrict__ b_off,
               float* __restrict__ py, float* __restrict__ px,
               float* __restrict__ msk) {
  __shared__ float wl[27 * 16 * 9];      // [o][cl][tap]
  const int g = blockIdx.x * 256 + threadIdx.x;
  const int b = g >> 12;
  const int h = (g >> 6) & 63;
  const int w = g & 63;

  float acc[27];
#pragma unroll
  for (int o = 0; o < 27; ++o) acc[o] = b_off[o];

  for (int cb = 0; cb < C_; cb += 16) {
    __syncthreads();
    for (int idx = threadIdx.x; idx < 27 * 16 * 9; idx += 256) {
      const int o = idx / 144, rem = idx % 144;
      const int cl = rem / 9, tap = rem % 9;
      wl[idx] = w_off[(o * C_ + (cb + cl)) * 9 + tap];
    }
    __syncthreads();
#pragma unroll 1
    for (int cl = 0; cl < 16; ++cl) {
      const int c = cb + cl;
      const float* xp = x + (size_t)(b * C_ + c) * HW_;
      float xv[9];
#pragma unroll
      for (int ky = 0; ky < 3; ++ky)
#pragma unroll
        for (int kx = 0; kx < 3; ++kx) {
          const int yy = h + ky - 1, xx = w + kx - 1;
          const bool ok = (yy >= 0) & (yy < H_) & (xx >= 0) & (xx < W_);
          xv[ky * 3 + kx] = ok ? xp[yy * W_ + xx] : 0.0f;
        }
      const float* wp = &wl[cl * 9];
#pragma unroll
      for (int o = 0; o < 27; ++o) {
        float s = 0.f;
#pragma unroll
        for (int tap = 0; tap < 9; ++tap) s = fmaf(xv[tap], wp[o * 144 + tap], s);
        acc[o] += s;
      }
    }
  }
#pragma unroll
  for (int kk = 0; kk < KK_; ++kk) {
    const int idx = (b * KK_ + kk) * HW_ + h * W_ + w;
    py[idx]  = (float)h + acc[kk];
    px[idx]  = (float)w + acc[9 + kk];
    msk[idx] = 1.0f / (1.0f + __expf(-acc[18 + kk]));
  }
}

// ---------------------------------------------------------------- kernel 3 --
// Block = one (b, h) row: 64 pixels x 256 out channels. 8 waves.
// Wave tile: all 4 m-tiles (64 px) x 2 n-tiles (32 out ch) -> 8 accumulators,
// each B fragment feeds 4 WMMAs; B ping-pong double-buffered across k-steps.
#define LDA 264   // bf16 row stride in LDS: 528 B = 33*16 -> b128-aligned rows

__global__ __launch_bounds__(256)
void k_fused(const float* __restrict__ x,
             const float* __restrict__ py, const float* __restrict__ px,
             const float* __restrict__ msk,
             const unsigned short* __restrict__ wb,   // bf16 [OC_][KD_]
             float* __restrict__ out) {
  __shared__ __align__(16) unsigned short As[64 * LDA];   // 33.8 KB

  const int tid  = threadIdx.x;
  const int bh   = blockIdx.x;
  const int b    = bh >> 6;
  const int h    = bh & 63;

  const int lane   = tid & 31;
  const int lane16 = lane & 15;
  const int hi     = lane >> 4;          // 0 | 1 (half-wave)
  const int wave   = tid >> 5;
  const int nbase  = wave * 32;          // this wave's 32 out channels

  const int p  = tid & 63;               // sampler: pixel (w coord)
  const int cg = tid >> 6;               // sampler: channel quarter (64 ch)

  union F { v16bf v; uint4 q[2]; };
  v8f acc[8] = {};                       // acc[mt*2 + nt]

  for (int kk = 0; kk < KK_; ++kk) {
    __syncthreads();                     // As consumers from prev kk are done

    // ---- bilinear sample (mask-premultiplied) -> bf16 A-tile in LDS ----
    const int   oidx = (b * KK_ + kk) * HW_ + h * W_ + p;
    const float fy = py[oidx], fx = px[oidx], mk = msk[oidx];
    const float y0f = floorf(fy), x0f = floorf(fx);
    const float wy = fy - y0f, wx = fx - x0f;
    const int y0 = (int)y0f, x0 = (int)x0f, y1 = y0 + 1, x1 = x0 + 1;
    const bool vy0 = (y0 >= 0) & (y0 < H_), vy1 = (y1 >= 0) & (y1 < H_);
    const bool vx0 = (x0 >= 0) & (x0 < W_), vx1 = (x1 >= 0) & (x1 < W_);
    float w00 = (vy0 & vx0) ? (1.f - wy) * (1.f - wx) * mk : 0.f;
    float w01 = (vy0 & vx1) ? (1.f - wy) * wx * mk : 0.f;
    float w10 = (vy1 & vx0) ? wy * (1.f - wx) * mk : 0.f;
    float w11 = (vy1 & vx1) ? wy * wx * mk : 0.f;
    const int yc0 = min(max(y0, 0), H_ - 1), yc1 = min(max(y1, 0), H_ - 1);
    const int xc0 = min(max(x0, 0), W_ - 1), xc1 = min(max(x1, 0), W_ - 1);
    const int i00 = yc0 * W_ + xc0, i01 = yc0 * W_ + xc1;
    const int i10 = yc1 * W_ + xc0, i11 = yc1 * W_ + xc1;

    unsigned int* As32 = reinterpret_cast<unsigned int*>(As);
    const float* xbase = x + (size_t)(b * C_ + cg * 64) * HW_;
#pragma unroll 1
    for (int i = 0; i < 32; ++i) {       // 2 channels per iter, packed b32
      const int cc = cg * 64 + 2 * i;
      const float* x0p = xbase + (size_t)(2 * i) * HW_;
      const float* x1p = x0p + HW_;
      const float v0 = w00 * x0p[i00] + w01 * x0p[i01] + w10 * x0p[i10] + w11 * x0p[i11];
      const float v1 = w00 * x1p[i00] + w01 * x1p[i01] + w10 * x1p[i10] + w11 * x1p[i11];
      As32[(p * LDA + cc) >> 1] = (unsigned)f2bf(v0) | ((unsigned)f2bf(v1) << 16);
    }
    __syncthreads();

    // ---- WMMA GEMM over this kk's 256 K-values ----
    // B lane layout: lane = column n, 16 contiguous K per lane (hi half +16).
    const unsigned short* wbase =
        wb + (size_t)(nbase + lane16) * KD_ + kk * 256 + hi * 16;
    // prefetch next kk's weight slab
    if (kk + 1 < KK_)
      __builtin_prefetch(wbase + 256, 0, 1);

    auto loadB = [&](F* dst, int ck) {
#pragma unroll
      for (int nt = 0; nt < 2; ++nt) {
        const uint4* bp =
            reinterpret_cast<const uint4*>(wbase + (size_t)nt * 16 * KD_ + ck);
        dst[nt].q[0] = bp[0];
        dst[nt].q[1] = bp[1];
      }
    };
    auto loadA = [&](F* a, int ck) {
#pragma unroll
      for (int mt = 0; mt < 4; ++mt) {
        const uint4* ap = reinterpret_cast<const uint4*>(
            &As[(mt * 16 + lane16) * LDA + ck + (hi ? 16 : 0)]);
        a[mt].q[0] = ap[0];       // K 0..7   (or 16..23 for hi half)
        a[mt].q[1] = ap[1];       // K 8..15  (or 24..31 for hi half)
      }
    };
    auto mm8 = [&](F* a, F* bb) {
#pragma unroll
      for (int nt = 0; nt < 2; ++nt)
#pragma unroll
        for (int mt = 0; mt < 4; ++mt)
          acc[mt * 2 + nt] = __builtin_amdgcn_wmma_f32_16x16x32_bf16(
              false, a[mt].v, false, bb[nt].v, (short)0,
              acc[mt * 2 + nt], false, false);
    };

    F b0[2], b1[2];
    loadB(b0, 0);
#pragma unroll 1
    for (int ck = 0; ck < 256; ck += 64) {     // 2 k-steps per iter, ping-pong
      F a0[4], a1[4];
      loadA(a0, ck);
      loadB(b1, ck + 32);                      // fill pong while ping computes
      mm8(a0, b0);
      loadA(a1, ck + 32);
      if (ck + 64 < 256) loadB(b0, ck + 64);   // refill ping
      mm8(a1, b1);
    }
  }

  // ---- epilogue: lane holds out-ch n fixed, 8 consecutive w pixels ----
#pragma unroll
  for (int mt = 0; mt < 4; ++mt)
#pragma unroll
    for (int nt = 0; nt < 2; ++nt) {
      const int n = nbase + nt * 16 + lane16;
      float* op = out + (size_t)((b * OC_ + n) * H_ + h) * W_ + mt * 16 + hi * 8;
      union { v8f v; float4 f4[2]; } r; r.v = acc[mt * 2 + nt];
      *reinterpret_cast<float4*>(op)     = r.f4[0];
      *reinterpret_cast<float4*>(op + 4) = r.f4[1];
    }
}

// ------------------------------------------------------------------ launch --
extern "C" void kernel_launch(void* const* d_in, const int* in_sizes, int n_in,
                              void* d_out, int out_size, void* d_ws, size_t ws_size,
                              hipStream_t stream) {
  (void)in_sizes; (void)n_in; (void)out_size; (void)ws_size;
  const float* x      = (const float*)d_in[0];
  const float* w_off  = (const float*)d_in[1];
  const float* b_off  = (const float*)d_in[2];
  const float* w_conv = (const float*)d_in[3];
  float* out = (float*)d_out;

  // workspace layout (needs ~4.72 MB):
  float* pyw = (float*)d_ws;
  float* pxw = pyw + NPKK;
  float* mw  = pxw + NPKK;
  unsigned short* wb = (unsigned short*)(mw + NPKK);   // bf16 [OC_][KD_]

  const int nw = OC_ * KD_;                            // 589824
  k_wcvt<<<(nw + 255) / 256, 256, 0, stream>>>(w_conv, wb, nw);
  k_offsets<<<(B_ * HW_) / 256, 256, 0, stream>>>(x, w_off, b_off, pyw, pxw, mw);
  k_fused<<<B_ * H_, 256, 0, stream>>>(x, pyw, pxw, mw, wb, out);
}
